// CondConv_59261958750578
// MI455X (gfx1250) — matrix operand order repaired and checked by reading, not compiled, not run on hardware
//
#include <hip/hip_runtime.h>
#include <hip/hip_bf16.h>

// MI455X / gfx1250 CondConv.
// Compute-bound conv (38.7 GFLOP) -> bf16 WMMA (v_wmma_f32_16x16x32_bf16),
// fp32 accumulation. Data movement uses the CDNA5 async-to-LDS engine:
// x is pre-converted to bf16 NHWC so the per-block input tile is staged with
// global_load_async_to_lds_b128 (ASYNCcnt, double-buffered, overlapping WMMA).
// agg_w is mixed into ws as bf16, tap-major K layout [b][dy*3+dx][o][c] so each
// K=32 WMMA slice is one tap x 32 contiguous channels.
// ws usage: 1 KiB (att) + 36 MiB (agg_w bf16) + 16 MiB (x bf16 NHWC) ~= 52 MB.

typedef __attribute__((ext_vector_type(16))) __bf16 v16bf;
typedef __attribute__((ext_vector_type(8)))  __bf16 v8bf;
typedef __attribute__((ext_vector_type(8)))  float  v8f;

static __device__ inline v16bf cat8(v8bf lo, v8bf hi) {
    v16bf r;
#pragma unroll
    for (int i = 0; i < 8; ++i) { r[i] = lo[i]; r[i + 8] = hi[i]; }
    return r;
}

// ---------------------------------------------------------------------------
// Kernel A: attention head. One block (256 thr = 8 waves) per sample.
// ---------------------------------------------------------------------------
__global__ __launch_bounds__(256) void attention_kernel(
    const float* __restrict__ x, const float* __restrict__ z,
    const float* __restrict__ w1, const float* __restrict__ gamma,
    const float* __restrict__ beta, const float* __restrict__ mean,
    const float* __restrict__ var, const float* __restrict__ w2,
    float* __restrict__ att)
{
    const int b    = blockIdx.x;
    const int tid  = threadIdx.x;
    const int lane = tid & 31;
    const int wave = tid >> 5;

    __shared__ float a[512];   // [z(256) | avg(256)]
    __shared__ float h[256];

    const float* xb = x + (size_t)b * 256 * 1024;
    for (int c = wave; c < 256; c += 8) {
        float s = 0.f;
        const float* xc = xb + (size_t)c * 1024;
        for (int i = lane; i < 1024; i += 32) s += xc[i];
#pragma unroll
        for (int o = 16; o > 0; o >>= 1) s += __shfl_down(s, o, 32);
        if (lane == 0) a[256 + c] = s * (1.0f / 1024.0f);
    }
    a[tid] = z[b * 256 + tid];
    __syncthreads();

    {
        const int t = tid;
        const float* w = w1 + (size_t)t * 512;
        float s = 0.f;
#pragma unroll 4
        for (int i = 0; i < 512; ++i) s += a[i] * w[i];
        s = (s - mean[t]) * rsqrtf(var[t] + 1e-5f) * gamma[t] + beta[t];
        h[t] = s > 0.f ? s : 0.f;
    }
    __syncthreads();

    if (tid < 4) {
        const float* w = w2 + tid * 256;
        float s = 0.f;
#pragma unroll 4
        for (int i = 0; i < 256; ++i) s += h[i] * w[i];
        att[b * 4 + tid] = 1.0f / (1.0f + __expf(-s));
    }
}

// ---------------------------------------------------------------------------
// Kernel B: x fp32 NCHW -> bf16 NHWC ws copy (channel-fastest so async-to-LDS
// 16B chunks are 8 consecutive channels). One block per (b, y) row.
// ---------------------------------------------------------------------------
__global__ __launch_bounds__(256) void xcvt_kernel(
    const float* __restrict__ x, __bf16* __restrict__ xw)
{
    const int b = blockIdx.x >> 5;
    const int y = blockIdx.x & 31;
    const int tid = threadIdx.x;
    __shared__ __bf16 ls[32 * 264];   // [xcol][c], padded row to dodge conflicts

    for (int idx = tid; idx < 8192; idx += 256) {       // read coalesced (NCHW)
        const int c  = idx >> 5;
        const int xc = idx & 31;
        ls[xc * 264 + c] =
            (__bf16)x[(((size_t)b * 256 + c) << 10) + (y << 5) + xc];
    }
    __syncthreads();
    for (int idx = tid; idx < 8192; idx += 256) {       // write coalesced (NHWC)
        const int xc = idx >> 8;
        const int c  = idx & 255;
        xw[(((size_t)b * 32 + y) * 32 + xc) * 256 + c] = ls[xc * 264 + c];
    }
}

// ---------------------------------------------------------------------------
// Kernel C: agg_w[b,t,o,c] = bf16( 0.1 * sum_k att[b,k]*weight[k,o,c,t]
//                                  + static_weight[o,c,t] ),  t = dy*3+dx.
// ---------------------------------------------------------------------------
__global__ __launch_bounds__(256) void mix_kernel(
    const float* __restrict__ weight, const float* __restrict__ sweight,
    const float* __restrict__ att, __bf16* __restrict__ aggw)
{
    const size_t total = (size_t)32 * 9 * 256 * 256;
    for (size_t i = (size_t)blockIdx.x * blockDim.x + threadIdx.x; i < total;
         i += (size_t)gridDim.x * blockDim.x) {
        const int c  = (int)(i & 255);
        const int o  = (int)((i >> 8) & 255);
        const int bt = (int)(i >> 16);
        const int t  = bt % 9;
        const int b  = bt / 9;
        const size_t wofs = ((size_t)o * 256 + c) * 9 + t;
        float v = sweight[wofs];
        const float* ab = att + b * 4;
#pragma unroll
        for (int k = 0; k < 4; ++k)
            v += 0.1f * ab[k] * weight[(size_t)k * (256 * 256 * 9) + wofs];
        aggw[i] = (__bf16)v;
    }
}

// ---------------------------------------------------------------------------
// Kernel D: per-sample implicit GEMM  out[o,p] = A[o,K] x B[K,p]
//   M=256 (o), N=1024 (p=y*32+x), K=2304 = 9 taps x 256 ch.
// Block tile 128(M) x 64(N), 8 wave32s, each wave 2x2 WMMA f32 accumulators.
// x tile (4 rows x 34 cols halo x 32 ch) double-buffered in LDS, filled with
// global_load_async_to_lds_b128 (3 uniform chunks/thread/group, OOB/padding
// chunks redirected to a trash slot so s_wait_asynccnt counts stay exact).
// ---------------------------------------------------------------------------
__global__ __launch_bounds__(256) void condconv_gemm(
    const __bf16* __restrict__ xw, const __bf16* __restrict__ aggw,
    float* __restrict__ out)
{
    const int ntile = blockIdx.x;        // 0..15  -> 64 pixels = 2 output rows
    const int mtile = blockIdx.y;        // 0..1   -> 128 output channels
    const int b     = blockIdx.z;        // 0..31

    const int tid  = threadIdx.x;
    const int wave = tid >> 5;
    const int lane = tid & 31;
    const int mi   = wave & 3;           // wave M position (4 x 32 rows)
    const int ni   = wave >> 2;          // wave N position (2 x 32 pixels)
    const int g    = lane >> 4;          // half-wave group (WMMA K split)
    const int ln   = lane & 15;

    const int y0     = ntile * 2;
    const int p0     = ntile * 64;
    const int blockM = mtile * 128;

    __shared__ __bf16 xs[2][4 * 34 * 32];          // [buf][r][colx][cc]
    __shared__ __align__(16) __bf16 trash_ls[8];   // sink for OOB/pad chunks

    const unsigned xsbase   = (unsigned)(uintptr_t)&xs[0][0];
    const unsigned trashoff = (unsigned)(uintptr_t)&trash_ls[0];
    const __bf16* xwb = xw + (size_t)b * (32 * 32 * 256);

    // Zero both buffers once (halo slots stay zero; async never writes them).
    {
        unsigned* w = (unsigned*)&xs[0][0];
        for (int i = tid; i < 4352; i += 256) w[i] = 0u;
    }

    // Issue one channel-group's tile copy: 544 live 16B chunks + pad -> 3/thread.
    auto issue_cg = [&](int cg, int p) {
#pragma unroll
        for (int k = 0; k < 3; ++k) {
            const int q    = tid + (k << 8);          // 0..767
            const int h    = q & 3;                   // 16B chunk within 32 ch
            const int rc   = q >> 2;
            const int colx = rc % 34;
            const int r    = rc / 34;
            const int row  = y0 - 1 + r;
            const int col  = colx - 1;
            const bool ok  = (q < 544) && (row >= 0) && (row < 32) &&
                             (col >= 0) && (col < 32);
            const __bf16* src =
                ok ? (xwb + ((((row << 5) + col) << 8) + (cg << 5) + (h << 3)))
                   : xwb;
            const unsigned ldst =
                ok ? (xsbase + (unsigned)p * 8704u + (unsigned)q * 16u)
                   : trashoff;
            asm volatile("global_load_async_to_lds_b128 %0, %1, off"
                         :: "v"(ldst), "v"(src) : "memory");
        }
    };

    v8f acc[2][2];
#pragma unroll
    for (int ms = 0; ms < 2; ++ms)
#pragma unroll
        for (int ns = 0; ns < 2; ++ns)
            acc[ms][ns] = (v8f){0.f, 0.f, 0.f, 0.f, 0.f, 0.f, 0.f, 0.f};

    __syncthreads();                 // zero-fill visible before first copy
    issue_cg(0, 0);

    for (int cg = 0; cg < 8; ++cg) {
        if (cg < 7) {
            issue_cg(cg + 1, (cg + 1) & 1);
            asm volatile("s_wait_asynccnt 3" ::: "memory");   // cg's 3 landed
        } else {
            asm volatile("s_wait_asynccnt 0" ::: "memory");
        }
        __syncthreads();             // all waves' copies for cg visible

        const __bf16* xsp = &xs[cg & 1][0];
        // Prefetch next channel group of this wave's A rows into GL2.
        if (cg < 7) {
            const int mpf = blockM + mi * 32 + ln;
            __builtin_prefetch(
                aggw + (((size_t)b * 9) * 256 + mpf) * 256 + (cg + 1) * 32, 0, 1);
        }

#pragma unroll
        for (int t = 0; t < 9; ++t) {    // 3x3 taps; K slice = 32 channels
            const int dy = t / 3, dx = t % 3;

            // A frags: contiguous K order used here (same as B):
            // lane<16 -> K {0..15}; lane>=16 -> K {16..31}.
            v16bf afrag[2];
#pragma unroll
            for (int s = 0; s < 2; ++s) {
                const int m = blockM + mi * 32 + s * 16 + ln;
                const __bf16* ap =
                    aggw + (((size_t)b * 9 + t) * 256 + m) * 256 + cg * 32;
                afrag[s] = *(const v16bf*)(ap + 16 * g);
            }

            // B frags: same contiguous K order as A: lanes 0-15 col n K=0..15,
            // lanes 16-31 col n K=16..31 -> 32 contiguous LDS bytes.
            v16bf bfrag[2];
#pragma unroll
            for (int s = 0; s < 2; ++s) {
                const int pofs = ni * 32 + s * 16 + ln;   // 0..63
                const int lr   = pofs >> 5;
                const int col  = pofs & 31;
                const int r    = lr + dy;                 // LDS row 0..3
                const int colx = col + dx;                // LDS col 0..33
                bfrag[s] = *(const v16bf*)&xsp[(r * 34 + colx) * 32 + 16 * g];
            }

#pragma unroll
            for (int ms = 0; ms < 2; ++ms)
#pragma unroll
                for (int ns = 0; ns < 2; ++ns)
                    acc[ms][ns] = __builtin_amdgcn_wmma_f32_16x16x32_bf16(
                        false, afrag[ms], false, bfrag[ns],
                        (short)0, acc[ms][ns], false, false);
        }
        __syncthreads();             // done reading buf before cg+2 overwrites
    }

    // Epilogue: C/D layout: VGPR r -> M = r + 8*(lane>>4), N = lane&15.
#pragma unroll
    for (int ms = 0; ms < 2; ++ms)
#pragma unroll
        for (int ns = 0; ns < 2; ++ns) {
            const int p = p0 + ni * 32 + ns * 16 + ln;
#pragma unroll
            for (int r = 0; r < 8; ++r) {
                const int m = blockM + mi * 32 + ms * 16 + 8 * g + r;
                out[(((size_t)b * 256 + m) << 10) + p] = acc[ms][ns][r];
            }
        }
}

// ---------------------------------------------------------------------------
extern "C" void kernel_launch(void* const* d_in, const int* in_sizes, int n_in,
                              void* d_out, int out_size, void* d_ws, size_t ws_size,
                              hipStream_t stream) {
    const float* x       = (const float*)d_in[0];
    const float* z       = (const float*)d_in[1];
    const float* w1      = (const float*)d_in[2];
    const float* gamma   = (const float*)d_in[3];
    const float* beta    = (const float*)d_in[4];
    const float* mean    = (const float*)d_in[5];
    const float* var     = (const float*)d_in[6];
    const float* w2      = (const float*)d_in[7];
    const float* weight  = (const float*)d_in[8];
    const float* sweight = (const float*)d_in[9];

    float*  att  = (float*)d_ws;
    __bf16* aggw = (__bf16*)((char*)d_ws + 1024);
    __bf16* xw   = (__bf16*)((char*)d_ws + 1024 + (size_t)37748736);

    attention_kernel<<<32, 256, 0, stream>>>(x, z, w1, gamma, beta, mean, var,
                                             w2, att);
    xcvt_kernel<<<1024, 256, 0, stream>>>(x, xw);
    mix_kernel<<<2048, 256, 0, stream>>>(weight, sweight, att, aggw);

    dim3 grid(16, 2, 32);   // (N tiles, M tiles, batch)
    condconv_gemm<<<grid, 256, 0, stream>>>(xw, aggw, (float*)d_out);
}
